// ECC_472446403145
// MI455X (gfx1250) — compile-verified, matrix operand order checked by reading, not run on hardware
//
#include <hip/hip_runtime.h>

// Problem sizes (match reference)
#define N_NODES 16384
#define N_EDGES 131072
#define NB      16
#define EDIM    4
#define HW      32
#define C1      24
#define C2      48
#define F1      64
#define NCLS    40

#define KCH 25          // K chunks of 32: 24 outer-product chunks + 1 bias tail (K = 800)
#define NT  3           // N tiles of 16 (C2 = 48)
#define BS_HALVES (KCH * NT * 32 * 16)   // 38400 f16 = 76800 bytes
#define XS_STRIDE 26                     // per-lane x strip stride (conflict-free)

typedef __attribute__((ext_vector_type(16))) _Float16 v16h;
typedef __attribute__((ext_vector_type(8)))  float    v8f;
typedef int async_v4i __attribute__((vector_size(16)));   // 128-bit payload type for async LDS DMA

#if defined(__gfx1250__) && __has_builtin(__builtin_amdgcn_global_load_async_to_lds_b128)
#define USE_ASYNC_LDS 1
#else
#define USE_ASYNC_LDS 0
#endif

// ---------------------------------------------------------------------------
// generic zero-fill (graph-capture-safe scratch init)
// ---------------------------------------------------------------------------
__global__ void ecc_zero(float* __restrict__ p, int n) {
    int i = blockIdx.x * blockDim.x + threadIdx.x;
    for (; i < n; i += gridDim.x * blockDim.x) p[i] = 0.0f;
}

// ---------------------------------------------------------------------------
// Build the swizzled f16 B image for the layer-2 WMMA GEMM (once per launch).
// Row kk of the virtual [800 x 48] B:  kk<768 -> fw2[(kk&31)*1152 + (kk>>5)*48 + o]
//                                      768<=kk<792 -> fb2[(kk-768)*48 + o], else 0.
// Stored pre-swizzled into per-lane WMMA B-operand order:
//   q = ((c*3 + nt)*32 + lane)*16 + t,  K-local(t,lanegrp) = (t<8?t:t+8) + 8*(lane>>4)
// ---------------------------------------------------------------------------
__global__ __launch_bounds__(256)
void ecc_prepB(const float* __restrict__ fw2, const float* __restrict__ fb2,
               _Float16* __restrict__ bsg)
{
    const int q = blockIdx.x * 256 + threadIdx.x;
    if (q >= BS_HALVES) return;
    const int t    = q & 15;
    const int lane = (q >> 4) & 31;
    const int nt   = (q >> 9) % NT;
    const int c    = q / (NT * 512);
    const int lg   = lane >> 4;
    const int base = (t < 8) ? t : t + 8;
    const int kk   = 32 * c + base + 8 * lg;
    const int col  = nt * 16 + (lane & 15);
    float v = 0.0f;
    if (kk < HW * C1) {
        const int k = kk & 31, i = kk >> 5;
        v = fw2[k * (C1 * C2) + i * C2 + col];
    } else if (kk < HW * C1 + C1) {
        v = fb2[(kk - HW * C1) * C2 + col];
    }
    bsg[q] = (_Float16)v;
}

// ---------------------------------------------------------------------------
// Layer 1: per-edge fnet (4->32->24), msg = x[src] * w, atomic scatter + degree
// ---------------------------------------------------------------------------
__global__ __launch_bounds__(256)
void ecc_layer1(const float* __restrict__ x, const int* __restrict__ ei,
                const float* __restrict__ ea,
                const float* __restrict__ fw1, const float* __restrict__ fb1,
                const float* __restrict__ fw2, const float* __restrict__ fb2,
                float* __restrict__ agg1, float* __restrict__ deg)
{
    __shared__ float sW1[EDIM * HW];
    __shared__ float sB1[HW];
    __shared__ float sW2[HW * C1];
    __shared__ float sB2[C1];
    for (int q = threadIdx.x; q < EDIM * HW; q += 256) sW1[q] = fw1[q];
    for (int q = threadIdx.x; q < HW;        q += 256) sB1[q] = fb1[q];
    for (int q = threadIdx.x; q < HW * C1;   q += 256) sW2[q] = fw2[q];
    for (int q = threadIdx.x; q < C1;        q += 256) sB2[q] = fb2[q];
    __syncthreads();

    const int e = blockIdx.x * 256 + threadIdx.x;
    if (e >= N_EDGES) return;

    const float4 a4 = *reinterpret_cast<const float4*>(ea + 4 * e);
    const int src = ei[e];
    const int dst = ei[N_EDGES + e];
    const float xs = x[src];

    float acc[C1];
    #pragma unroll
    for (int i = 0; i < C1; ++i) acc[i] = sB2[i];

    #pragma unroll 4
    for (int j = 0; j < HW; ++j) {
        float h = sB1[j] + a4.x * sW1[0 * HW + j] + a4.y * sW1[1 * HW + j]
                         + a4.z * sW1[2 * HW + j] + a4.w * sW1[3 * HW + j];
        h = h > 0.0f ? h : 0.0f;
        #pragma unroll
        for (int i = 0; i < C1; ++i) acc[i] += h * sW2[j * C1 + i];
    }
    #pragma unroll
    for (int i = 0; i < C1; ++i) atomicAdd(&agg1[dst * C1 + i], xs * acc[i]);
    atomicAdd(&deg[dst], 1.0f);
}

// ---------------------------------------------------------------------------
// Node update 1: h1 = ELU(x*root1 + agg1/deg + bias1); also per-graph counts
// ---------------------------------------------------------------------------
__global__ __launch_bounds__(256)
void ecc_node1(const float* __restrict__ x, const float* __restrict__ agg1,
               const float* __restrict__ deg,
               const float* __restrict__ root1, const float* __restrict__ bias1,
               const int* __restrict__ batch,
               float* __restrict__ h1, float* __restrict__ gcnt)
{
    const int n = blockIdx.x * 256 + threadIdx.x;
    if (n >= N_NODES) return;
    const float xv = x[n];
    const float invd = 1.0f / fmaxf(deg[n], 1.0f);
    #pragma unroll
    for (int i = 0; i < C1; ++i) {
        float v = xv * root1[i] + agg1[n * C1 + i] * invd + bias1[i];
        h1[n * C1 + i] = v > 0.0f ? v : (__expf(v) - 1.0f);
    }
    atomicAdd(&gcnt[batch[n]], 1.0f);
}

// ---------------------------------------------------------------------------
// Layer 2 (dominant GEMM, WMMA): per 16-edge tile,
//   msg = [outer(h_e, x_e) | x_e] @ [fw2 | fb2]  (K = 792, padded to 800)
// B staged into LDS once per block via async DMA (ASYNCcnt path); K-chunk loop
// kept as a runtime loop so B fragments stream from LDS (no spill-hoisting).
// ---------------------------------------------------------------------------
__global__ __launch_bounds__(256, 1)
void ecc_layer2_wmma(const float* __restrict__ edge_attr,
                     const int*   __restrict__ edge_index,
                     const float* __restrict__ h1,
                     const float* __restrict__ fw1, const float* __restrict__ fb1,
                     const _Float16* __restrict__ bsg,
                     float* __restrict__ agg2)
{
    __shared__ __attribute__((aligned(32))) _Float16 Bs[BS_HALVES];      // 75 KB
    __shared__ _Float16 Xs[8 * 32 * XS_STRIDE];                          // 13 KB
    __shared__ float Wf[EDIM * HW + HW];                                 // fnet2 w1|b1

    const int tid = threadIdx.x;

    for (int q = tid; q < EDIM * HW + HW; q += 256)
        Wf[q] = (q < EDIM * HW) ? fw1[q] : fb1[q - EDIM * HW];

    // ---- stage B image into LDS --------------------------------------------
#if USE_ASYNC_LDS
    {
        const char* gsrc = (const char*)bsg;
        char*       ldst = (char*)&Bs[0];
        for (int off = tid * 16; off < BS_HALVES * 2; off += 256 * 16) {
            __builtin_amdgcn_global_load_async_to_lds_b128(
                (__attribute__((address_space(1))) async_v4i*)(gsrc + off),
                (__attribute__((address_space(3))) async_v4i*)(ldst + off),
                0, 0);
        }
#if __has_builtin(__builtin_amdgcn_s_wait_asynccnt)
        __builtin_amdgcn_s_wait_asynccnt(0);
#else
        asm volatile("s_wait_asynccnt 0" ::: "memory");
#endif
    }
#else
    {
        const uint4* gsrc = (const uint4*)bsg;
        uint4*       ldst = (uint4*)&Bs[0];
        for (int q = tid; q < BS_HALVES / 8; q += 256) ldst[q] = gsrc[q];
    }
#endif
    __syncthreads();

    const int lane = tid & 31;
    const int wave = tid >> 5;
    const int lg   = lane >> 4;
    const int el   = lane & 15;
    const int xbase = (wave * 32 + lane) * XS_STRIDE;
    const int tstride = gridDim.x * 8;

    for (int tile = blockIdx.x * 8 + wave; tile < N_EDGES / 16; tile += tstride) {
        const int e = tile * 16 + el;

        // speculative prefetch of next tile's edge attrs (global_prefetch_b8)
        if (tile + tstride < N_EDGES / 16)
            __builtin_prefetch(edge_attr + 4 * (e + 16 * tstride), 0, 1);

        const float4 a4 = *reinterpret_cast<const float4*>(edge_attr + 4 * e);
        const int src = edge_index[e];

        // fnet2 hidden: hv[j] = relu(b1[j] + ea . W1[:,j])
        float hv[HW];
        #pragma unroll
        for (int j = 0; j < HW; ++j) {
            float s = Wf[EDIM * HW + j]
                    + a4.x * Wf[0 * HW + j] + a4.y * Wf[1 * HW + j]
                    + a4.z * Wf[2 * HW + j] + a4.w * Wf[3 * HW + j];
            hv[j] = s > 0.0f ? s : 0.0f;
        }

        // per-lane K-local hidden values (compile-time indices, one lg-select)
        _Float16 hl[16];
        #pragma unroll
        for (int t = 0; t < 16; ++t) {
            const int b0 = (t < 8) ? t : t + 8;
            hl[t] = (_Float16)(lg ? hv[b0 + 8] : hv[b0]);
        }

        // source-node features (h1 row): keep in regs for the tail, and stage a
        // per-lane LDS strip so the runtime chunk loop can index it.
        _Float16 xh[C1];
        const float* xr = h1 + src * C1;
        #pragma unroll
        for (int i = 0; i < C1; i += 4) {
            const float4 v4 = *reinterpret_cast<const float4*>(xr + i);
            xh[i + 0] = (_Float16)v4.x; xh[i + 1] = (_Float16)v4.y;
            xh[i + 2] = (_Float16)v4.z; xh[i + 3] = (_Float16)v4.w;
        }
        #pragma unroll
        for (int i = 0; i < C1; ++i) Xs[xbase + i] = xh[i];

        v8f acc0 = {}, acc1 = {}, acc2 = {};

        // 24 outer-product K-chunks: A[t] = hl[t] * x[c].
        // Runtime loop (unroll 2): B fragments stream from LDS each iteration.
        #pragma unroll 2
        for (int c = 0; c < C1; ++c) {
            const _Float16 xc = Xs[xbase + c];
            v16h a;
            #pragma unroll
            for (int t = 0; t < 16; ++t) a[t] = hl[t] * xc;
            const v16h b0 = *reinterpret_cast<const v16h*>(&Bs[(c * NT + 0) * 512 + lane * 16]);
            const v16h b1 = *reinterpret_cast<const v16h*>(&Bs[(c * NT + 1) * 512 + lane * 16]);
            const v16h b2 = *reinterpret_cast<const v16h*>(&Bs[(c * NT + 2) * 512 + lane * 16]);
            acc0 = __builtin_amdgcn_wmma_f32_16x16x32_f16(false, a, false, b0, (short)0, acc0, false, false);
            acc1 = __builtin_amdgcn_wmma_f32_16x16x32_f16(false, a, false, b1, (short)0, acc1, false, false);
            acc2 = __builtin_amdgcn_wmma_f32_16x16x32_f16(false, a, false, b2, (short)0, acc2, false, false);
        }

        // tail chunk (bias rows): A = x passthrough (K-local 24..31 are zero)
        {
            v16h a;
            #pragma unroll
            for (int t = 0; t < 16; ++t) {
                const int b0 = (t < 8) ? t : t + 8;
                const _Float16 v0 = (b0 < C1)     ? xh[b0]     : (_Float16)0.0f;
                const _Float16 v1 = (b0 + 8 < C1) ? xh[b0 + 8] : (_Float16)0.0f;
                a[t] = lg ? v1 : v0;
            }
            const v16h b0 = *reinterpret_cast<const v16h*>(&Bs[(C1 * NT + 0) * 512 + lane * 16]);
            const v16h b1 = *reinterpret_cast<const v16h*>(&Bs[(C1 * NT + 1) * 512 + lane * 16]);
            const v16h b2 = *reinterpret_cast<const v16h*>(&Bs[(C1 * NT + 2) * 512 + lane * 16]);
            acc0 = __builtin_amdgcn_wmma_f32_16x16x32_f16(false, a, false, b0, (short)0, acc0, false, false);
            acc1 = __builtin_amdgcn_wmma_f32_16x16x32_f16(false, a, false, b1, (short)0, acc1, false, false);
            acc2 = __builtin_amdgcn_wmma_f32_16x16x32_f16(false, a, false, b2, (short)0, acc2, false, false);
        }

        // scatter: C/D layout -> lane<16: M=r, lane>=16: M=r+8; N = (lane&15)+16*nt
        const int ebase = tile * 16 + 8 * lg;
        #pragma unroll
        for (int r = 0; r < 8; ++r) {
            const int dst = edge_index[N_EDGES + ebase + r];
            float* out = agg2 + dst * C2 + el;
            atomicAdd(out +  0, acc0[r]);
            atomicAdd(out + 16, acc1[r]);
            atomicAdd(out + 32, acc2[r]);
        }
    }
}

// ---------------------------------------------------------------------------
// Node update 2: h2 = ELU(h1@root2 + agg2/deg + bias2), fused mean-pool accum
// ---------------------------------------------------------------------------
__global__ __launch_bounds__(256)
void ecc_node2(const float* __restrict__ h1, const float* __restrict__ agg2,
               const float* __restrict__ deg,
               const float* __restrict__ root2, const float* __restrict__ bias2,
               const int* __restrict__ batch, float* __restrict__ gsum)
{
    const int idx = blockIdx.x * 256 + threadIdx.x;
    if (idx >= N_NODES * C2) return;
    const int n = idx / C2, c = idx % C2;
    float s = bias2[c] + agg2[idx] / fmaxf(deg[n], 1.0f);
    const float* hr = h1 + n * C1;
    #pragma unroll
    for (int i = 0; i < C1; ++i) s += hr[i] * root2[i * C2 + c];
    const float v = s > 0.0f ? s : (__expf(s) - 1.0f);
    atomicAdd(&gsum[batch[n] * C2 + c], v);
}

// ---------------------------------------------------------------------------
// Head: mean pool -> fc1+ELU -> fc2 -> log_softmax  (tiny: single block)
// ---------------------------------------------------------------------------
__global__ __launch_bounds__(256)
void ecc_head(const float* __restrict__ gsum, const float* __restrict__ gcnt,
              const float* __restrict__ fc1w, const float* __restrict__ fc1b,
              const float* __restrict__ fc2w, const float* __restrict__ fc2b,
              float* __restrict__ out)
{
    __shared__ float g[NB * C2];
    __shared__ float z[NB * F1];
    __shared__ float lgt[NB * NCLS];
    const int tid = threadIdx.x;

    for (int q = tid; q < NB * C2; q += 256) {
        const int b = q / C2;
        g[q] = gsum[q] / fmaxf(gcnt[b], 1.0f);
    }
    __syncthreads();
    for (int q = tid; q < NB * F1; q += 256) {
        const int b = q / F1, f = q % F1;
        float s = fc1b[f];
        for (int c = 0; c < C2; ++c) s += g[b * C2 + c] * fc1w[c * F1 + f];
        z[q] = s > 0.0f ? s : (__expf(s) - 1.0f);
    }
    __syncthreads();
    for (int q = tid; q < NB * NCLS; q += 256) {
        const int b = q / NCLS, k = q % NCLS;
        float s = fc2b[k];
        for (int f = 0; f < F1; ++f) s += z[b * F1 + f] * fc2w[f * NCLS + k];
        lgt[q] = s;
    }
    __syncthreads();
    if (tid < NB) {
        float m = -1e30f;
        for (int k = 0; k < NCLS; ++k) m = fmaxf(m, lgt[tid * NCLS + k]);
        float s = 0.0f;
        for (int k = 0; k < NCLS; ++k) s += __expf(lgt[tid * NCLS + k] - m);
        const float ls = __logf(s) + m;
        for (int k = 0; k < NCLS; ++k) out[tid * NCLS + k] = lgt[tid * NCLS + k] - ls;
    }
}

// ---------------------------------------------------------------------------
extern "C" void kernel_launch(void* const* d_in, const int* in_sizes, int n_in,
                              void* d_out, int out_size, void* d_ws, size_t ws_size,
                              hipStream_t stream)
{
    const float* x     = (const float*)d_in[0];
    const int*   ei    = (const int*)  d_in[1];
    const float* ea    = (const float*)d_in[2];
    const int*   batch = (const int*)  d_in[3];
    const float* f1w1  = (const float*)d_in[4];
    const float* f1b1  = (const float*)d_in[5];
    const float* f1w2  = (const float*)d_in[6];
    const float* f1b2  = (const float*)d_in[7];
    const float* root1 = (const float*)d_in[8];
    const float* bias1 = (const float*)d_in[9];
    const float* f2w1  = (const float*)d_in[10];
    const float* f2b1  = (const float*)d_in[11];
    const float* f2w2  = (const float*)d_in[12];
    const float* f2b2  = (const float*)d_in[13];
    const float* root2 = (const float*)d_in[14];
    const float* bias2 = (const float*)d_in[15];
    const float* fc1w  = (const float*)d_in[16];
    const float* fc1b  = (const float*)d_in[17];
    const float* fc2w  = (const float*)d_in[18];
    const float* fc2b  = (const float*)d_in[19];

    float* agg1 = (float*)d_ws;                 // N*C1
    float* deg  = agg1 + N_NODES * C1;          // N
    float* h1   = deg  + N_NODES;               // N*C1
    float* agg2 = h1   + N_NODES * C1;          // N*C2
    float* gsum = agg2 + N_NODES * C2;          // NB*C2
    float* gcnt = gsum + NB * C2;               // NB
    _Float16* bsg = (_Float16*)(gcnt + NB);     // BS_HALVES (16B-aligned offset)

    // zero the accumulators (agg1+deg contiguous; agg2+gsum+gcnt contiguous)
    ecc_zero<<<256, 256, 0, stream>>>(agg1, N_NODES * C1 + N_NODES);
    ecc_zero<<<256, 256, 0, stream>>>(agg2, N_NODES * C2 + NB * C2 + NB);

    ecc_prepB <<<(BS_HALVES + 255) / 256, 256, 0, stream>>>(f2w2, f2b2, bsg);
    ecc_layer1<<<N_EDGES / 256, 256, 0, stream>>>(x, ei, ea, f1w1, f1b1, f1w2, f1b2, agg1, deg);
    ecc_node1 <<<N_NODES / 256, 256, 0, stream>>>(x, agg1, deg, root1, bias1, batch, h1, gcnt);
    ecc_layer2_wmma<<<128, 256, 0, stream>>>(ea, ei, h1, f2w1, f2b1, bsg, agg2);
    ecc_node2 <<<(N_NODES * C2) / 256, 256, 0, stream>>>(h1, agg2, deg, root2, bias2, batch, gsum);
    ecc_head  <<<1, 256, 0, stream>>>(gsum, gcnt, fc1w, fc1b, fc2w, fc2b, (float*)d_out);
}
